// SLoRA_50405736186182
// MI455X (gfx1250) — matrix operand order repaired and checked by reading, not compiled, not run on hardware
//
#include <hip/hip_runtime.h>
#include <hip/hip_bf16.h>

#define T_TOK   4096
#define DIN     2048
#define DOUT    2048
#define NA      8
#define MR      64
#define NSLOTS  (NA * MR)
#define MAX_TILES 264

typedef __attribute__((ext_vector_type(16))) __bf16 v16bf;
typedef __attribute__((ext_vector_type(8)))  __bf16 v8bf;
typedef __attribute__((ext_vector_type(8)))  float  v8f;

__device__ __forceinline__ v16bf cat16(v8bf lo, v8bf hi) {
    return __builtin_shufflevector(lo, hi, 0,1,2,3,4,5,6,7,8,9,10,11,12,13,14,15);
}

// A-operand: 16 fp32 -> v16bf. Lane layout (ISA 7.12.2, 16-bit A 16x32):
// elements 0..7  = K = kbase+{0..7}, elements 8..15 = K = kbase+16+{0..7}.
__device__ __forceinline__ v16bf loadA16(const float* __restrict__ p) {
    float f[16];
    *(float4*)&f[0]  = *(const float4*)(p);
    *(float4*)&f[4]  = *(const float4*)(p + 4);
    *(float4*)&f[8]  = *(const float4*)(p + 16);
    *(float4*)&f[12] = *(const float4*)(p + 20);
    v16bf r;
#pragma unroll
    for (int i = 0; i < 16; ++i) r[i] = (__bf16)f[i];
    return r;
}

// B-operand: 16 bf16 (lane = column, K = hi*8+{0..7}, 16+hi*8+{0..7}).
__device__ __forceinline__ v16bf loadB16(const __bf16* __restrict__ q) {
    return cat16(*(const v8bf*)q, *(const v8bf*)(q + 16));
}

// ---------------------------------------------------------------------------
// Kernel 1: bin tokens by adapter, build tile descriptors.
// ws_i layout (int32):
//   [0..7]   seg_start[a]
//   [8..15]  count[a]
//   [16]     n_tiles
//   [32+2i]  tile adapter,  [33+2i] tile absolute offset into tok_sorted
//   [1024..1024+T) tok_sorted
// ---------------------------------------------------------------------------
__global__ __launch_bounds__(256) void slora_bin_kernel(
    const int* __restrict__ adapter_ids, int* __restrict__ ws_i)
{
    __shared__ int cnt[NA], cur[NA];
    const int tid = threadIdx.x;
    if (tid < NA) cnt[tid] = 0;
    __syncthreads();
    for (int t = tid; t < T_TOK; t += 256)
        atomicAdd(&cnt[adapter_ids[t]], 1);
    __syncthreads();
    if (tid == 0) {
        int off = 0, nt = 0;
        for (int a = 0; a < NA; ++a) {
            cur[a] = off;
            ws_i[a]     = off;
            ws_i[8 + a] = cnt[a];
            const int tiles = (cnt[a] + 15) >> 4;
            for (int ti = 0; ti < tiles; ++ti) {
                ws_i[32 + 2 * nt] = a;
                ws_i[33 + 2 * nt] = off + ti * 16;
                ++nt;
            }
            off += cnt[a];
        }
        ws_i[16] = nt;
    }
    __syncthreads();
    for (int t = tid; t < T_TOK; t += 256) {
        const int aid = adapter_ids[t];
        const int pos = atomicAdd(&cur[aid], 1);
        ws_i[1024 + pos] = t;
    }
}

// ---------------------------------------------------------------------------
// Kernel 2: resolve rank_offsets indirection; convert caches to bf16.
//   a_bf  [s=a*64+r][din]     (K-contiguous B-operand for GEMM1)
//   bt_bf [a][o][r]           (rank innermost => K-contiguous B-operand GEMM2)
// ---------------------------------------------------------------------------
__global__ __launch_bounds__(256) void slora_convert_kernel(
    const float* __restrict__ a_cache, const float* __restrict__ b_cache,
    const int* __restrict__ rank_offsets,
    __bf16* __restrict__ a_bf, __bf16* __restrict__ bt_bf)
{
    const int s    = blockIdx.x;           // logical slot
    const int slot = rank_offsets[s];      // physical page
    const int a = s >> 6, r = s & 63;
    const float* pa = a_cache + (size_t)slot * DIN;
    const float* pb = b_cache + (size_t)slot * DOUT;
    for (int c = threadIdx.x; c < DIN; c += 256)
        a_bf[(size_t)s * DIN + c] = (__bf16)pa[c];
    for (int c = threadIdx.x; c < DOUT; c += 256)
        bt_bf[((size_t)a * DOUT + c) * MR + r] = (__bf16)pb[c];
}

// ---------------------------------------------------------------------------
// Kernel 3: fused grouped-GEMM tile. 128 threads = 4 waves.
//  Stage 1: xa[16x64] = x_tile(16x2048) * A_a^T   (64 wmma per wave, pipelined)
//  Stage 2: out tile = base + scale * xa * B_a    (2 wmma per N-tile)
// ---------------------------------------------------------------------------
__global__ __launch_bounds__(128) void slora_tile_kernel(
    const float* __restrict__ x, const float* __restrict__ base,
    const int* __restrict__ ranks, const float* __restrict__ scalings,
    const int* __restrict__ ws_i,
    const __bf16* __restrict__ a_bf, const __bf16* __restrict__ bt_bf,
    float* __restrict__ out)
{
    __shared__ int    tok_s[16];
    __shared__ __bf16 xa_lds[16 * MR];

    const int n_tiles = ws_i[16];
    const int tile = blockIdx.x;
    if (tile >= n_tiles) return;

    const int a   = ws_i[32 + 2 * tile];
    const int off = ws_i[33 + 2 * tile];
    const int seg_end = ws_i[a] + ws_i[8 + a];

    const int tid = threadIdx.x;
    if (tid < 16) {
        const int gi = off + tid;
        tok_s[tid] = (gi < seg_end) ? ws_i[1024 + gi] : -1;
    }
    __syncthreads();

    int rank = ranks[a];
    rank = rank < 0 ? 0 : (rank > MR ? MR : rank);
    const float scal = scalings[a];

    const int w   = tid >> 5;       // wave id 0..3
    const int l   = tid & 31;       // lane
    const int row = l & 15;         // M-row / N-col within 16
    const int hi  = l >> 4;         // half-wave

    // ---------------- Stage 1: xa = x * A^T (branch-free, pipelined) -------
    const int  n      = w * 16 + row;        // rank column supplied by this lane
    const bool bvalid = (n < rank);          // applied once at the LDS store
    const int  tok    = tok_s[row];
    const int  tok_eff = (tok >= 0) ? tok : 0;  // dead rows compute garbage,
                                                // never stored (finite data)
    const float*  p = x    + (size_t)tok_eff * DIN + hi * 8;
    const __bf16* q = a_bf + (size_t)(a * MR + n) * DIN + hi * 8;

    v8f c = {};
    v16bf aop = loadA16(p);
    v16bf bop = loadB16(q);
#pragma unroll 2
    for (int k0 = 0; k0 < DIN - 32; k0 += 32) {
        p += 32; q += 32;
        __builtin_prefetch((const void*)(p + 512), 0, 1);
        v16bf an = loadA16(p);           // next iteration's operands in flight
        v16bf bn = loadB16(q);
        c = __builtin_amdgcn_wmma_f32_16x16x32_bf16(
                false, aop, false, bop, (short)0, c, false, false);
        aop = an; bop = bn;
    }
    c = __builtin_amdgcn_wmma_f32_16x16x32_bf16(
            false, aop, false, bop, (short)0, c, false, false);

    // D layout: lane holds N = row, M = hi*8 + v. Rank-mask applied here.
#pragma unroll
    for (int v = 0; v < 8; ++v)
        xa_lds[(hi * 8 + v) * MR + n] = bvalid ? (__bf16)c[v] : (__bf16)0.0f;
    __syncthreads();

    // ---------------- Stage 2: y = xa * B, out = base + scale*y ------------
    v16bf a2[2];
    {
        const __bf16* prow = &xa_lds[row * MR];
#pragma unroll
        for (int ks = 0; ks < 2; ++ks) {
            const int k0 = ks * 32;
            v8bf c0 = *(const v8bf*)(prow + k0 + hi * 8);
            v8bf c1 = *(const v8bf*)(prow + k0 + 16 + hi * 8);
            a2[ks] = cat16(c0, c1);
        }
    }

    const __bf16* pbase_bt = bt_bf + (size_t)a * DOUT * MR;
    for (int nt = w; nt < DOUT / 16; nt += 4) {
        const int o = nt * 16 + row;                  // output column this lane
        const __bf16* pbo = pbase_bt + (size_t)o * MR;
        v8f acc = {};
#pragma unroll
        for (int ks = 0; ks < 2; ++ks) {
            v16bf bop = loadB16(pbo + ks * 32 + hi * 8);
            acc = __builtin_amdgcn_wmma_f32_16x16x32_bf16(
                      false, a2[ks], false, bop, (short)0, acc, false, false);
        }
#pragma unroll
        for (int v = 0; v < 8; ++v) {
            const int m = hi * 8 + v;
            const int t = tok_s[m];
            if (t >= 0) {
                const size_t idx = (size_t)t * DOUT + o;
                out[idx] = base[idx] + scal * acc[v];
            }
        }
    }
}

// ---------------------------------------------------------------------------
extern "C" void kernel_launch(void* const* d_in, const int* in_sizes, int n_in,
                              void* d_out, int out_size, void* d_ws, size_t ws_size,
                              hipStream_t stream) {
    const float* x            = (const float*)d_in[0];
    const float* base_output  = (const float*)d_in[1];
    const int*   adapter_ids  = (const int*)d_in[2];
    const int*   ranks        = (const int*)d_in[3];
    const float* scalings     = (const float*)d_in[4];
    const int*   rank_offsets = (const int*)d_in[5];
    const float* a_cache      = (const float*)d_in[6];
    const float* b_cache      = (const float*)d_in[7];
    float*       out          = (float*)d_out;

    char* ws = (char*)d_ws;
    int*    ws_i  = (int*)ws;                                      // 32 KB ints
    __bf16* a_bf  = (__bf16*)(ws + (32 << 10));                    // 2 MB
    __bf16* bt_bf = (__bf16*)(ws + (32 << 10) + (size_t)NSLOTS * DIN * 2); // 2 MB

    slora_bin_kernel<<<1, 256, 0, stream>>>(adapter_ids, ws_i);
    slora_convert_kernel<<<NSLOTS, 256, 0, stream>>>(a_cache, b_cache,
                                                     rank_offsets, a_bf, bt_bf);
    slora_tile_kernel<<<MAX_TILES, 128, 0, stream>>>(x, base_output, ranks,
                                                     scalings, ws_i, a_bf,
                                                     bt_bf, out);
}
